// TransformerBlock_83330955477142
// MI455X (gfx1250) — matrix operand order repaired and checked by reading, not compiled, not run on hardware
//
#include <hip/hip_runtime.h>

// ---------------------------------------------------------------------------
// Transformer block for MI455X (gfx1250, wave32, WMMA bf16 -> f32 accumulate)
// B=2 S=2048 D=1024 H=16 DH=64 M=4096.  Compute-bound (~137 GFLOP vs ~100MB),
// so all GEMM-shaped work is routed through v_wmma_f32_16x16x32_bf16, and
// tile staging uses global_load_async_to_lds (ASYNCcnt) when available.
// Workspace requirement: ~152 MB.
// ---------------------------------------------------------------------------

#define B_ 2
#define S_ 2048
#define D_ 1024
#define H_ 16
#define DH_ 64
#define M_ 4096
#define NROWS (B_ * S_) // 4096

typedef __bf16 v16bf __attribute__((ext_vector_type(16)));
typedef float v8f __attribute__((ext_vector_type(8)));
typedef unsigned short u16x8 __attribute__((ext_vector_type(8)));
typedef unsigned short u16x16 __attribute__((ext_vector_type(16)));
typedef unsigned short u16x4 __attribute__((ext_vector_type(4)));
typedef int gv4i __attribute__((vector_size(16))); // matches builtin's V4i

#define V8F_ZERO ((v8f){0.f, 0.f, 0.f, 0.f, 0.f, 0.f, 0.f, 0.f})

static __device__ __forceinline__ unsigned short f2bf(float f) {
  unsigned int u = __float_as_uint(f);
  u += 0x7fffu + ((u >> 16) & 1u); // round-to-nearest-even
  return (unsigned short)(u >> 16);
}

static __device__ __forceinline__ v16bf mk16(u16x8 lo, u16x8 hi) {
  u16x16 t = __builtin_shufflevector(lo, hi, 0, 1, 2, 3, 4, 5, 6, 7, 8, 9, 10,
                                     11, 12, 13, 14, 15);
  union { u16x16 u; v16bf b; } cv;
  cv.u = t;
  return cv.b;
}

static __device__ __forceinline__ v8f wmma_bf(v16bf a, v16bf b, v8f c) {
  // D = A(16x32 bf16) * B(32x16 bf16) + C(16x16 f32)
  return __builtin_amdgcn_wmma_f32_16x16x32_bf16(false, a, false, b, (short)0,
                                                 c, false, false);
}

// 16-byte global -> LDS copy. Prefers the CDNA5 async path (ASYNCcnt); falls
// back to a synchronous VGPR round-trip if the builtin is unavailable.
// Low 32 bits of a generic shared-aperture pointer are the LDS byte offset.
static __device__ __forceinline__ void async_g2l_b128(
    const unsigned short* g, unsigned short* l) {
#if defined(__gfx1250__) && \
    __has_builtin(__builtin_amdgcn_global_load_async_to_lds_b128)
  __builtin_amdgcn_global_load_async_to_lds_b128(
      (__attribute__((address_space(1))) gv4i*)(uintptr_t)g,
      (__attribute__((address_space(3))) gv4i*)(unsigned int)(uintptr_t)l,
      0, 0);
#else
  *(u16x8*)l = *(const u16x8*)g;
#endif
}

static __device__ __forceinline__ void wait_async() {
#if defined(__gfx1250__) && __has_builtin(__builtin_amdgcn_s_wait_asynccnt)
  __builtin_amdgcn_s_wait_asynccnt(0);
#elif defined(__HIP_DEVICE_COMPILE__)
  asm volatile("s_wait_asynccnt 0" ::: "memory");
#endif
}

static __device__ __forceinline__ float fast_tanh(float x) {
  // tanh(x) = 1 - 2/(exp(2x)+1), on fast-path __expf
  return 1.0f - 2.0f / (__expf(2.0f * x) + 1.0f);
}

// ---------------------------------------------------------------------------
// LayerNorm (fp32 in) fused with bf16 conversion. One block per row of D=1024.
// ---------------------------------------------------------------------------
__global__ __launch_bounds__(256) void ln_bf16_kernel(
    const float* __restrict__ x, const float* __restrict__ sc,
    const float* __restrict__ bi, unsigned short* __restrict__ out) {
  const int row = blockIdx.x;
  const int t = threadIdx.x;
  const float4 v = ((const float4*)(x + (size_t)row * D_))[t];
  float s = v.x + v.y + v.z + v.w;
  float s2 = v.x * v.x + v.y * v.y + v.z * v.z + v.w * v.w;
#pragma unroll
  for (int off = 16; off; off >>= 1) {
    s += __shfl_xor(s, off, 32);
    s2 += __shfl_xor(s2, off, 32);
  }
  __shared__ float ps[8], ps2[8], stat[2];
  if ((t & 31) == 0) { ps[t >> 5] = s; ps2[t >> 5] = s2; }
  __syncthreads();
  if (t == 0) {
    float ts = 0.f, ts2 = 0.f;
#pragma unroll
    for (int i = 0; i < 8; i++) { ts += ps[i]; ts2 += ps2[i]; }
    float mean = ts * (1.0f / D_);
    float var = ts2 * (1.0f / D_) - mean * mean;
    stat[0] = mean;
    stat[1] = rsqrtf(var + 1e-6f);
  }
  __syncthreads();
  const float mean = stat[0], rstd = stat[1];
  const float4 scv = ((const float4*)sc)[t];
  const float4 biv = ((const float4*)bi)[t];
  u16x4 o;
  o[0] = f2bf((v.x - mean) * rstd * scv.x + biv.x);
  o[1] = f2bf((v.y - mean) * rstd * scv.y + biv.y);
  o[2] = f2bf((v.z - mean) * rstd * scv.z + biv.z);
  o[3] = f2bf((v.w - mean) * rstd * scv.w + biv.w);
  ((u16x4*)(out + (size_t)row * D_))[t] = o;
}

// ---------------------------------------------------------------------------
// Transpose + fp32->bf16 weight conversion: src[R][C] -> dst[C][R] bf16.
// ---------------------------------------------------------------------------
__global__ __launch_bounds__(256) void wcvt_t_kernel(
    const float* __restrict__ src, unsigned short* __restrict__ dst, int R,
    int C) {
  __shared__ float tile[32][33];
  const int tx = threadIdx.x & 31, ty = threadIdx.x >> 5;
  const int r0 = blockIdx.y * 32, c0 = blockIdx.x * 32;
#pragma unroll
  for (int j = 0; j < 4; j++) {
    int r = ty + j * 8;
    tile[r][tx] = src[(size_t)(r0 + r) * C + c0 + tx];
  }
  __syncthreads();
#pragma unroll
  for (int j = 0; j < 4; j++) {
    int n = ty + j * 8;
    dst[(size_t)(c0 + n) * R + r0 + tx] = f2bf(tile[tx][n]);
  }
}

// ---------------------------------------------------------------------------
// Tiled bf16 GEMM: out = A[Mr][K] * Bt[N][K]^T, f32 accumulate.
// Block = 128x128 tile, 8 waves (4x2), wave = 32x64 (2x4 WMMA tiles), BK=64.
// Staging via async global->LDS; 16 WMMAs per barrier pair.
// Epilogue: optional bias, tanh-GELU, fp32 residual; writes f32 and/or bf16.
// ---------------------------------------------------------------------------
__global__ __launch_bounds__(256) void gemm_bf16_kernel(
    const unsigned short* __restrict__ A, const unsigned short* __restrict__ Bt,
    float* __restrict__ outF, unsigned short* __restrict__ outBf,
    const float* __restrict__ bias, const float* __restrict__ resid, int K,
    int Ncols, int apply_gelu) {
  __shared__ unsigned short la[128][72]; // [m][k], pad -> 16B-aligned,
  __shared__ unsigned short lb[128][72]; // [n][k], conflict-free frag reads
  const int t = threadIdx.x;
  const int lane = t & 31, wave = t >> 5;
  const int wm = wave >> 1, wn = wave & 1;
  const int half = lane >> 4, l16 = lane & 15;
  const int mBase = blockIdx.y * 128, nBase = blockIdx.x * 128;

  v8f acc[2][4];
#pragma unroll
  for (int i = 0; i < 2; i++)
#pragma unroll
    for (int j = 0; j < 4; j++) acc[i][j] = V8F_ZERO;

  for (int k0 = 0; k0 < K; k0 += 64) {
    __syncthreads();
#pragma unroll
    for (int i = 0; i < 4; i++) {
      int c = t + i * 256; // 1024 chunks of 8 bf16 per matrix
      int row = c >> 3, off = (c & 7) * 8;
      async_g2l_b128(A + (size_t)(mBase + row) * K + k0 + off, &la[row][off]);
      async_g2l_b128(Bt + (size_t)(nBase + row) * K + k0 + off, &lb[row][off]);
    }
    if (k0 + 64 < K) { // gfx1250 global_prefetch of next K tile
      int row = t >> 1, off = (t & 1) * 32;
      __builtin_prefetch(A + (size_t)(mBase + row) * K + k0 + 64 + off, 0, 1);
      __builtin_prefetch(Bt + (size_t)(nBase + row) * K + k0 + 64 + off, 0, 1);
    }
    wait_async();
    __syncthreads();

#pragma unroll
    for (int ks = 0; ks < 2; ks++) { // two 32-K substeps
      const int kb = ks * 32;
      v16bf af[2], bfm[4];
#pragma unroll
      for (int mi = 0; mi < 2; mi++) { // A frag: lane=M, K {h8+0..7,16+h8+..}
        int r = wm * 32 + mi * 16 + l16;
        u16x8 lo = *(const u16x8*)&la[r][kb + half * 8];
        u16x8 hi = *(const u16x8*)&la[r][kb + 16 + half * 8];
        af[mi] = mk16(lo, hi);
      }
#pragma unroll
      for (int ni = 0; ni < 4; ni++) { // B frag: lane=N, K {h8+0..7,16+h8+..}
        int n = wn * 64 + ni * 16 + l16;
        u16x8 lo = *(const u16x8*)&lb[n][kb + half * 8];
        u16x8 hi = *(const u16x8*)&lb[n][kb + 16 + half * 8];
        bfm[ni] = mk16(lo, hi);
      }
#pragma unroll
      for (int mi = 0; mi < 2; mi++)
#pragma unroll
        for (int ni = 0; ni < 4; ni++)
          acc[mi][ni] = wmma_bf(af[mi], bfm[ni], acc[mi][ni]);
    }
  }

  // Epilogue. C layout: VGPR r -> M = r + 8*half, col = lane&15.
#pragma unroll
  for (int mi = 0; mi < 2; mi++) {
#pragma unroll
    for (int ni = 0; ni < 4; ni++) {
      const int col = nBase + wn * 64 + ni * 16 + l16;
      const float bv = bias ? bias[col] : 0.0f;
#pragma unroll
      for (int r = 0; r < 8; r++) {
        const int row = mBase + wm * 32 + mi * 16 + r + half * 8;
        float v = acc[mi][ni][r] + bv;
        if (apply_gelu) {
          float x3 = v * v * v;
          v = 0.5f * v *
              (1.0f + fast_tanh(0.7978845608028654f * (v + 0.044715f * x3)));
        }
        if (resid) v += resid[(size_t)row * Ncols + col];
        if (outF) outF[(size_t)row * Ncols + col] = v;
        if (outBf) outBf[(size_t)row * Ncols + col] = f2bf(v);
      }
    }
  }
}

// ---------------------------------------------------------------------------
// RoPE on Q,K (fp32 in) -> bf16 out; folds 1/sqrt(DH)=0.125 into Q.
// ---------------------------------------------------------------------------
__global__ __launch_bounds__(256) void rope_kernel(
    const float* __restrict__ Q, const float* __restrict__ Kk,
    unsigned short* __restrict__ Qb, unsigned short* __restrict__ Kb) {
  const int idx = blockIdx.x * 256 + threadIdx.x; // (b,s,h)*32 + pair
  const int p = idx & 31;
  const int s = (idx >> 9) & (S_ - 1);
  const float freq = __powf(10000.0f, -(float)p * (1.0f / 32.0f));
  float sn, cs;
  __sincosf((float)s * freq, &sn, &cs);
  const size_t base = (size_t)(idx >> 5) * DH_ + 2 * p;
  float a = Q[base], b = Q[base + 1];
  Qb[base] = f2bf((a * cs - b * sn) * 0.125f);
  Qb[base + 1] = f2bf((a * sn + b * cs) * 0.125f);
  a = Kk[base];
  b = Kk[base + 1];
  Kb[base] = f2bf(a * cs - b * sn);
  Kb[base + 1] = f2bf(a * sn + b * cs);
}

// ---------------------------------------------------------------------------
// Flash attention. One wave per 16-row Q tile; 8 waves/block share LDS-staged
// K / V^T 32-row KV tiles (K staged async). scores: 4 WMMAs, P*V: 4 WMMAs per
// KV tile. Online softmax via 16-lane shfl_xor reductions; P re-layout
// (C-layout -> A-layout) through wave-private LDS with s_wait_dscnt.
// ---------------------------------------------------------------------------
__global__ __launch_bounds__(256) void attn_kernel(
    const unsigned short* __restrict__ Qb, const unsigned short* __restrict__ Kb,
    const unsigned short* __restrict__ Vb, unsigned short* __restrict__ ctx) {
  __shared__ unsigned short lk[32][72];     // [kv][d]   (pad: conflict-free)
  __shared__ unsigned short lvt[64][40];    // [d][kv]   (V transposed)
  __shared__ unsigned short lp[8][16][40];  // per-wave P tile [m][kv]
  const int t = threadIdx.x, lane = t & 31, wave = t >> 5;
  const int half = lane >> 4, l16 = lane & 15;
  const int bh = blockIdx.x >> 4, qgrp = blockIdx.x & 15;
  const int bb = bh >> 4, h = bh & (H_ - 1);
  const int q0 = (qgrp * 8 + wave) * 16;

  // Q A-fragments for K-dim = DH = 64 (2 frags of 16x32)
  const size_t qrow = ((size_t)(bb * S_ + q0 + l16) * H_ + h) * DH_;
  v16bf qf[2];
#pragma unroll
  for (int f = 0; f < 2; f++) {
    u16x8 lo = *(const u16x8*)(Qb + qrow + f * 32 + half * 8);
    u16x8 hi = *(const u16x8*)(Qb + qrow + f * 32 + 16 + half * 8);
    qf[f] = mk16(lo, hi);
  }

  float m_i[8], l_i[8];
  v8f o[4];
#pragma unroll
  for (int r = 0; r < 8; r++) { m_i[r] = -1e30f; l_i[r] = 0.0f; }
#pragma unroll
  for (int dt = 0; dt < 4; dt++) o[dt] = V8F_ZERO;

  for (int kv0 = 0; kv0 < S_; kv0 += 32) {
    __syncthreads();
    { // stage K async; stage V with in-flight transpose
      const int row = t >> 3, ch = t & 7;
      const size_t g = ((size_t)(bb * S_ + kv0 + row) * H_ + h) * DH_ + ch * 8;
      async_g2l_b128(Kb + g, &lk[row][ch * 8]);
      u16x8 vv = *(const u16x8*)(Vb + g);
#pragma unroll
      for (int j = 0; j < 8; j++) lvt[ch * 8 + j][row] = vv[j];
    }
    wait_async();
    __syncthreads();

    // scores S = Q * K^T : two 16x16 tiles (kv cols 0-15, 16-31)
    v8f s0 = V8F_ZERO, s1 = V8F_ZERO;
#pragma unroll
    for (int nt = 0; nt < 2; nt++) {
      const int kvc = nt * 16 + l16; // B-frag col (kv), rows = d
      v8f acc = nt ? s1 : s0;
#pragma unroll
      for (int f = 0; f < 2; f++) { // dBase = f*32
        u16x8 lo = *(const u16x8*)&lk[kvc][f * 32 + half * 8];
        u16x8 hi = *(const u16x8*)&lk[kvc][f * 32 + 16 + half * 8];
        acc = wmma_bf(qf[f], mk16(lo, hi), acc);
      }
      if (nt) s1 = acc; else s0 = acc;
    }

    // online softmax (row r lives at M = r + 8*half, 16 lanes hold 16 cols)
    float sc[8];
#pragma unroll
    for (int r = 0; r < 8; r++) {
      float v = fmaxf(s0[r], s1[r]);
      v = fmaxf(v, __shfl_xor(v, 1, 32));
      v = fmaxf(v, __shfl_xor(v, 2, 32));
      v = fmaxf(v, __shfl_xor(v, 4, 32));
      v = fmaxf(v, __shfl_xor(v, 8, 32));
      const float mn = fmaxf(m_i[r], v);
      sc[r] = __expf(m_i[r] - mn);
      const float p0 = __expf(s0[r] - mn);
      const float p1 = __expf(s1[r] - mn);
      const int m = r + half * 8;
      lp[wave][m][l16] = f2bf(p0);
      lp[wave][m][16 + l16] = f2bf(p1);
      float rs = p0 + p1;
      rs += __shfl_xor(rs, 1, 32);
      rs += __shfl_xor(rs, 2, 32);
      rs += __shfl_xor(rs, 4, 32);
      rs += __shfl_xor(rs, 8, 32);
      l_i[r] = l_i[r] * sc[r] + rs;
      m_i[r] = mn;
    }
#pragma unroll
    for (int dt = 0; dt < 4; dt++)
#pragma unroll
      for (int r = 0; r < 8; r++) o[dt][r] *= sc[r];

    // wave-local LDS fence before re-reading P in A-layout (CDNA5 counter)
    asm volatile("s_wait_dscnt 0" ::: "memory");

    u16x8 plo = *(const u16x8*)&lp[wave][l16][half * 8];
    u16x8 phi = *(const u16x8*)&lp[wave][l16][16 + half * 8];
    const v16bf pf = mk16(plo, phi);
#pragma unroll
    for (int dt = 0; dt < 4; dt++) { // ctx += P(16x32) * V(32x16)
      const int d = dt * 16 + l16;
      u16x8 lo = *(const u16x8*)&lvt[d][half * 8];
      u16x8 hi = *(const u16x8*)&lvt[d][16 + half * 8];
      o[dt] = wmma_bf(pf, mk16(lo, hi), o[dt]);
    }
  }

#pragma unroll
  for (int r = 0; r < 8; r++) l_i[r] = 1.0f / l_i[r];
#pragma unroll
  for (int dt = 0; dt < 4; dt++) {
#pragma unroll
    for (int r = 0; r < 8; r++) {
      const int m = r + half * 8;
      const size_t g =
          ((size_t)(bb * S_ + q0 + m) * H_ + h) * DH_ + dt * 16 + l16;
      ctx[g] = f2bf(o[dt][r] * l_i[r]);
    }
  }
}

// ---------------------------------------------------------------------------
extern "C" void kernel_launch(void* const* d_in, const int* in_sizes, int n_in,
                              void* d_out, int out_size, void* d_ws,
                              size_t ws_size, hipStream_t stream) {
  const float* x = (const float*)d_in[0];
  const float* ln1_s = (const float*)d_in[1];
  const float* ln1_b = (const float*)d_in[2];
  const float* wk = (const float*)d_in[3];
  const float* wq = (const float*)d_in[4];
  const float* wv = (const float*)d_in[5];
  const float* wo = (const float*)d_in[6];
  const float* ln2_s = (const float*)d_in[7];
  const float* ln2_b = (const float*)d_in[8];
  const float* w1 = (const float*)d_in[9];
  const float* b1 = (const float*)d_in[10];
  const float* w2 = (const float*)d_in[11];
  const float* b2 = (const float*)d_in[12];
  float* out = (float*)d_out;

  char* ws = (char*)d_ws;
  size_t off = 0;
  auto alloc = [&](size_t bytes) -> void* {
    void* p = ws + off;
    off += (bytes + 255) & ~(size_t)255;
    return p;
  };
  unsigned short* h1 = (unsigned short*)alloc((size_t)NROWS * D_ * 2);
  unsigned short* wqT = (unsigned short*)alloc((size_t)D_ * D_ * 2);
  unsigned short* wkT = (unsigned short*)alloc((size_t)D_ * D_ * 2);
  unsigned short* wvT = (unsigned short*)alloc((size_t)D_ * D_ * 2);
  unsigned short* woT = (unsigned short*)alloc((size_t)D_ * D_ * 2);
  unsigned short* w1T = (unsigned short*)alloc((size_t)D_ * M_ * 2);
  unsigned short* w2T = (unsigned short*)alloc((size_t)M_ * D_ * 2);
  float* Qf = (float*)alloc((size_t)NROWS * D_ * 4);
  float* Kf = (float*)alloc((size_t)NROWS * D_ * 4);
  unsigned short* Qbb = (unsigned short*)alloc((size_t)NROWS * D_ * 2);
  unsigned short* Kbb = (unsigned short*)alloc((size_t)NROWS * D_ * 2);
  unsigned short* Vbb = (unsigned short*)alloc((size_t)NROWS * D_ * 2);
  unsigned short* ctxb = (unsigned short*)alloc((size_t)NROWS * D_ * 2);
  float* mlp_in = (float*)alloc((size_t)NROWS * D_ * 4);
  unsigned short* h2 = (unsigned short*)alloc((size_t)NROWS * D_ * 2);
  unsigned short* ffb = (unsigned short*)alloc((size_t)NROWS * M_ * 2);
  (void)ws_size; // requires ~152 MB of workspace

  // 1) LN1 -> bf16
  ln_bf16_kernel<<<NROWS, 256, 0, stream>>>(x, ln1_s, ln1_b, h1);
  // 2) weight transpose+convert
  wcvt_t_kernel<<<dim3(32, 32), 256, 0, stream>>>(wq, wqT, D_, D_);
  wcvt_t_kernel<<<dim3(32, 32), 256, 0, stream>>>(wk, wkT, D_, D_);
  wcvt_t_kernel<<<dim3(32, 32), 256, 0, stream>>>(wv, wvT, D_, D_);
  wcvt_t_kernel<<<dim3(32, 32), 256, 0, stream>>>(wo, woT, D_, D_);
  wcvt_t_kernel<<<dim3(M_ / 32, D_ / 32), 256, 0, stream>>>(w1, w1T, D_, M_);
  wcvt_t_kernel<<<dim3(D_ / 32, M_ / 32), 256, 0, stream>>>(w2, w2T, M_, D_);
  // 3) QKV projections
  const dim3 gD(D_ / 128, NROWS / 128);
  gemm_bf16_kernel<<<gD, 256, 0, stream>>>(h1, wqT, Qf, nullptr, nullptr,
                                           nullptr, D_, D_, 0);
  gemm_bf16_kernel<<<gD, 256, 0, stream>>>(h1, wkT, Kf, nullptr, nullptr,
                                           nullptr, D_, D_, 0);
  gemm_bf16_kernel<<<gD, 256, 0, stream>>>(h1, wvT, nullptr, Vbb, nullptr,
                                           nullptr, D_, D_, 0);
  // 4) RoPE (+ fold score scale into Q), -> bf16
  rope_kernel<<<(B_ * S_ * H_ * (DH_ / 2)) / 256, 256, 0, stream>>>(Qf, Kf,
                                                                    Qbb, Kbb);
  // 5) attention
  attn_kernel<<<B_ * H_ * (S_ / 128), 256, 0, stream>>>(Qbb, Kbb, Vbb, ctxb);
  // 6) output projection + residual -> mlp_in (f32)
  gemm_bf16_kernel<<<gD, 256, 0, stream>>>(ctxb, woT, mlp_in, nullptr, nullptr,
                                           x, D_, D_, 0);
  // 7) LN2 -> bf16
  ln_bf16_kernel<<<NROWS, 256, 0, stream>>>(mlp_in, ln2_s, ln2_b, h2);
  // 8) MLP up + GELU -> bf16
  gemm_bf16_kernel<<<dim3(M_ / 128, NROWS / 128), 256, 0, stream>>>(
      h2, w1T, nullptr, ffb, b1, nullptr, D_, M_, 1);
  // 9) MLP down + bias + residual -> final f32 output
  gemm_bf16_kernel<<<gD, 256, 0, stream>>>(ffb, w2T, out, nullptr, b2, mlp_in,
                                           M_, D_, 0);
  (void)in_sizes; (void)n_in; (void)out_size;
}
